// MultiHeadedAttention_65077344469572
// MI455X (gfx1250) — matrix operand order repaired and checked by reading, not compiled, not run on hardware
//
#include <hip/hip_runtime.h>
#include <stdint.h>

// ---------------------------------------------------------------------------
// MI455X (gfx1250) multi-headed attention per the (buggy) reference:
//   pre-convert inputs fp32->bf16, pre-transpose weights to bf16 [N][K],
//   K/Q/V proj (3x GEMM 16384x1024x1024) -> per-(b,h) 128-row attention with
//   d=1024 -> output proj with Wv/bv.
// All GEMMs on v_wmma_f32_16x16x32_bf16. Tile staging uses double-buffered
// global_load_async_to_lds_b128 (ASYNCcnt): issue tile i+1, wait <=4 so only
// tile i must have landed, compute tile i while i+1 streams in.
// ---------------------------------------------------------------------------

typedef __attribute__((ext_vector_type(16))) __bf16 v16bf;
typedef __attribute__((ext_vector_type(8)))  __bf16 v8bf;
typedef __attribute__((ext_vector_type(8)))  float  v8f;

#define N_HEADS 16
#define D_MODEL 1024
#define BB 8
#define SS 2048
#define MTOT (BB * SS) /* 16384 rows */

// fp32 -> bf16 round-to-nearest-even
static __device__ __forceinline__ __bf16 f2bf(float f) {
  union { float f; uint32_t u; } v; v.f = f;
  uint32_t r = v.u + 0x7FFFu + ((v.u >> 16) & 1u);
  uint16_t h = (uint16_t)(r >> 16);
  __bf16 o; __builtin_memcpy(&o, &h, 2);
  return o;
}

// Async DMA: 16 bytes global -> LDS, tracked by ASYNCcnt (CDNA5).
static __device__ __forceinline__ void async_cp_b128(void* lds, const void* g) {
  unsigned lo = (unsigned)(uintptr_t)lds;               // LDS byte address
  unsigned long long ga = (unsigned long long)(uintptr_t)g;
  asm volatile("global_load_async_to_lds_b128 %0, %1, off"
               :: "v"(lo), "v"(ga) : "memory");
}
static __device__ __forceinline__ void wait_async0() {
  asm volatile("s_wait_asynccnt 0x0" ::: "memory");
}
// Async loads retire in order: <=4 outstanding means the older 4 (current
// tile) have landed while the newer 4 (next tile) may still be in flight.
static __device__ __forceinline__ void wait_async4() {
  asm volatile("s_wait_asynccnt 0x4" ::: "memory");
}

static __device__ __forceinline__ v16bf combine16(v8bf lo, v8bf hi) {
  v16bf r;
#pragma unroll
  for (int i = 0; i < 8; ++i) { r[i] = lo[i]; r[i + 8] = hi[i]; }
  return r;
}

// A fragment (16x32 bf16) from row-major LDS tile, row stride `stride` elems.
static __device__ __forceinline__ v16bf ld_a(const __bf16* base, int row, int stride, int khalf) {
  const __bf16* p = base + (size_t)row * stride;
  v8bf lo = *(const v8bf*)(p + khalf * 8);
  v8bf hi = *(const v8bf*)(p + 16 + khalf * 8);
  return combine16(lo, hi);
}

// B fragment (32x16 bf16) from N-major LDS tile [n][k], row stride `stride`.
static __device__ __forceinline__ v16bf ld_b(const __bf16* base, int col, int stride, int khalf) {
  const __bf16* p = base + (size_t)col * stride + khalf * 8;
  v8bf lo = *(const v8bf*)(p);
  v8bf hi = *(const v8bf*)(p + 16);
  return combine16(lo, hi);
}

static __device__ __forceinline__ v8f wmma_bf16(v16bf a, v16bf b, v8f c) {
  return __builtin_amdgcn_wmma_f32_16x16x32_bf16(
      /*neg_a=*/false, a, /*neg_b=*/false, b,
      /*c_mod=*/(short)0, c, /*reuse_a=*/false, /*reuse_b=*/false);
}

// ---------------------------------------------------------------------------
// Elementwise fp32 -> bf16 (n multiple of 2048)
// ---------------------------------------------------------------------------
__global__ __launch_bounds__(256) void cvt_bf16_kernel(
    const float* __restrict__ in, __bf16* __restrict__ out, int n)
{
  const int i = (blockIdx.x * 256 + threadIdx.x) * 8;
  if (i >= n) return;
  float4 a = ((const float4*)(in + i))[0];
  float4 b = ((const float4*)(in + i))[1];
  v8bf p;
  p[0] = f2bf(a.x); p[1] = f2bf(a.y); p[2] = f2bf(a.z); p[3] = f2bf(a.w);
  p[4] = f2bf(b.x); p[5] = f2bf(b.y); p[6] = f2bf(b.z); p[7] = f2bf(b.w);
  *(v8bf*)(out + i) = p;
}

// ---------------------------------------------------------------------------
// Weight pre-transpose: W fp32 [K][N] -> Wt bf16 [N][K]  (D x D, D=1024)
// ---------------------------------------------------------------------------
__global__ __launch_bounds__(256) void transpose_w_kernel(
    const float* __restrict__ W, __bf16* __restrict__ Wt, int D)
{
  __shared__ __bf16 Ls[32][33];
  const int t  = threadIdx.x;
  const int tx = t & 31, ty = t >> 5;                 // ty = 0..7
  const int x0 = blockIdx.x * 32, y0 = blockIdx.y * 32;
#pragma unroll
  for (int i = 0; i < 4; ++i)
    Ls[ty + 8 * i][tx] = f2bf(W[(size_t)(y0 + ty + 8 * i) * D + x0 + tx]);
  __syncthreads();
#pragma unroll
  for (int i = 0; i < 4; ++i)
    Wt[(size_t)(x0 + ty + 8 * i) * D + y0 + tx] = Ls[tx][ty + 8 * i];
}

// ---------------------------------------------------------------------------
// GEMM: C[M,N] = A[M,K] @ Bt[N,K]^T + bias[N]   (A, Bt bf16; Bt is N-major)
// Block tile 128x128, K-tile 32, 256 threads (8 waves), 2x4 wmma tiles/wave.
// LDS: 2x (As[128][40] + Bs[128][40]) bf16 = 40960 B, double-buffered async.
// Buffer addresses are computed arithmetically (no LDS-pointer arrays: those
// become addrspacecast static initializers that ld.lld rejects).
// ---------------------------------------------------------------------------
template <bool OUT_F32>
__global__ __launch_bounds__(256) void gemm_bias_bf16(
    const __bf16* __restrict__ A, const __bf16* __restrict__ Bt,
    const float* __restrict__ bias, void* __restrict__ Cp,
    int M, int N, int K)
{
  extern __shared__ char smem[];

  const int t     = threadIdx.x;
  const int lane  = t & 31, wave = t >> 5;
  const int waveM = wave & 3, waveN = wave >> 2;
  const int m0    = blockIdx.x * 128;
  const int n0    = blockIdx.y * 128;
  const int khalf = lane >> 4;
  const int lc    = lane & 15;

  auto AsOf = [&](int buf) -> __bf16* { return (__bf16*)(smem + buf * 20480); };
  auto BsOf = [&](int buf) -> __bf16* { return (__bf16*)(smem + 10240 + buf * 20480); };

  v8f acc[2][4];
#pragma unroll
  for (int mt = 0; mt < 2; ++mt)
#pragma unroll
    for (int nt = 0; nt < 4; ++nt)
#pragma unroll
      for (int j = 0; j < 8; ++j) acc[mt][nt][j] = 0.0f;

  const int srow = t >> 1, scb = (t & 1) * 16;  // 16 elems (32 B) per thread per tile

  auto issue = [&](int kt, int buf) {
    const __bf16* ag = A  + (size_t)(m0 + srow) * K + kt + scb;
    const __bf16* bg = Bt + (size_t)(n0 + srow) * K + kt + scb;
    __bf16* as = AsOf(buf);
    __bf16* bs = BsOf(buf);
    async_cp_b128(as + srow * 40 + scb,     ag);
    async_cp_b128(as + srow * 40 + scb + 8, ag + 8);
    async_cp_b128(bs + srow * 40 + scb,     bg);
    async_cp_b128(bs + srow * 40 + scb + 8, bg + 8);
  };

  const int T = K / 32;
  issue(0, 0);
  for (int i = 0; i < T; ++i) {
    const int cur = i & 1;
    if (i + 1 < T) { issue((i + 1) * 32, cur ^ 1); wait_async4(); }
    else           { wait_async0(); }
    __syncthreads();

    const __bf16* As = AsOf(cur);
    const __bf16* Bs = BsOf(cur);
    v16bf af[2], bfv[4];
#pragma unroll
    for (int mt = 0; mt < 2; ++mt)
      af[mt] = ld_a(As, waveM * 32 + mt * 16 + lc, 40, khalf);
#pragma unroll
    for (int nt = 0; nt < 4; ++nt)
      bfv[nt] = ld_b(Bs, waveN * 64 + nt * 16 + lc, 40, khalf);
#pragma unroll
    for (int mt = 0; mt < 2; ++mt)
#pragma unroll
      for (int nt = 0; nt < 4; ++nt)
        acc[mt][nt] = wmma_bf16(af[mt], bfv[nt], acc[mt][nt]);
    __syncthreads();   // all waves done reading `cur` before it is refilled
  }

  // epilogue: bias add + store (C layout: VGPR j -> M = j + 8*khalf, N = lc)
#pragma unroll
  for (int nt = 0; nt < 4; ++nt) {
    const int ncol = n0 + waveN * 64 + nt * 16 + lc;
    const float bv_ = bias[ncol];
#pragma unroll
    for (int mt = 0; mt < 2; ++mt) {
#pragma unroll
      for (int j = 0; j < 8; ++j) {
        const int r = m0 + waveM * 32 + mt * 16 + j + 8 * khalf;
        const float val = acc[mt][nt][j] + bv_;
        if (OUT_F32) ((float*)Cp)[(size_t)r * N + ncol] = val;
        else         ((__bf16*)Cp)[(size_t)r * N + ncol] = f2bf(val);
      }
    }
  }
}

// ---------------------------------------------------------------------------
// Attention: one workgroup per (b,h) chunk of 128 rows, d = 1024.
//   S = Q K^T / 32 (WMMA, double-buffered async staging) -> row softmax ->
//   O = P V (WMMA).
// Dynamic LDS: 2x(As+Bs)[128][40] (40960) + Sf[128][128]f32 @40960 (65536,
// overlaid by Vs[128][136]bf16 after softmax) + Pb[128][136]bf16 @106496
// = 141312 B total.
// ---------------------------------------------------------------------------
__global__ __launch_bounds__(256) void attn_kernel(
    const __bf16* __restrict__ Qp, const __bf16* __restrict__ Kp,
    const __bf16* __restrict__ Vp, __bf16* __restrict__ Op)
{
  extern __shared__ char smem[];

  const int bh = blockIdx.x;              // b*16 + h ; chunk rows = bh*128
  const size_t rowBase = (size_t)bh * 128;

  const int t     = threadIdx.x;
  const int lane  = t & 31, wave = t >> 5;
  const int khalf = lane >> 4, lc = lane & 15;

  auto AsOf = [&](int buf) -> __bf16* { return (__bf16*)(smem + buf * 20480); };
  auto BsOf = [&](int buf) -> __bf16* { return (__bf16*)(smem + 10240 + buf * 20480); };
  float*  Sf = (float*)(smem + 40960);    // scores [128][128] fp32
  __bf16* Vs = (__bf16*)(smem + 40960);   // overlay: V^T tile [128][136]
  __bf16* Pb = (__bf16*)(smem + 106496);  // P bf16 [128][136]

  // ---- S = Q K^T ---- (each wave: 16 rows x full 128 cols = 8 tiles)
  v8f acc[8];
#pragma unroll
  for (int nt = 0; nt < 8; ++nt)
#pragma unroll
    for (int j = 0; j < 8; ++j) acc[nt][j] = 0.0f;

  const int srow = t >> 1, scb = (t & 1) * 16;
  auto issueQK = [&](int kt, int buf) {
    const __bf16* q = Qp + (rowBase + srow) * D_MODEL + kt + scb;
    const __bf16* k = Kp + (rowBase + srow) * D_MODEL + kt + scb;
    __bf16* as = AsOf(buf);
    __bf16* bs = BsOf(buf);
    async_cp_b128(as + srow * 40 + scb,     q);
    async_cp_b128(as + srow * 40 + scb + 8, q + 8);
    async_cp_b128(bs + srow * 40 + scb,     k);
    async_cp_b128(bs + srow * 40 + scb + 8, k + 8);
  };

  const int T = D_MODEL / 32;  // 32
  issueQK(0, 0);
  for (int i = 0; i < T; ++i) {
    const int cur = i & 1;
    if (i + 1 < T) { issueQK((i + 1) * 32, cur ^ 1); wait_async4(); }
    else           { wait_async0(); }
    __syncthreads();

    v16bf a = ld_a(AsOf(cur), wave * 16 + lc, 40, khalf);
#pragma unroll
    for (int nt = 0; nt < 8; ++nt) {
      v16bf b = ld_b(BsOf(cur), nt * 16 + lc, 40, khalf);
      acc[nt] = wmma_bf16(a, b, acc[nt]);
    }
    __syncthreads();
  }

  // scaled scores -> LDS
  const float scale = 0.03125f; // 1/sqrt(d_model=1024)
#pragma unroll
  for (int nt = 0; nt < 8; ++nt)
#pragma unroll
    for (int j = 0; j < 8; ++j) {
      const int r = wave * 16 + j + 8 * khalf;
      Sf[r * 128 + nt * 16 + lc] = acc[nt][j] * scale;
    }
  __syncthreads();

  // ---- softmax: thread t owns row t ----
  if (t < 128) {
    float mx = -3.0e38f;
    for (int c = 0; c < 128; ++c) mx = fmaxf(mx, Sf[t * 128 + c]);
    float sum = 0.0f;
    for (int c = 0; c < 128; ++c) {
      float e = __expf(Sf[t * 128 + c] - mx);
      Sf[t * 128 + c] = e;
      sum += e;
    }
    const float inv = 1.0f / sum;
    for (int c = 0; c < 128; ++c) Pb[t * 136 + c] = f2bf(Sf[t * 128 + c] * inv);
  }
  __syncthreads();

  // ---- O = P V : K = 128 ; loop over 8 output n-tiles of 128 cols ----
  const int vk = t >> 1, vnb = (t & 1) * 64;
  for (int ntile = 0; ntile < 8; ++ntile) {
    // stage V^T tile into Vs[n][k] (overlays Sf, safe after barrier)
#pragma unroll
    for (int c = 0; c < 8; ++c) {
      v8bf chunk = *(const v8bf*)(Vp + (rowBase + vk) * D_MODEL + ntile * 128 + vnb + c * 8);
#pragma unroll
      for (int e = 0; e < 8; ++e)
        Vs[(size_t)(vnb + c * 8 + e) * 136 + vk] = chunk[e];
    }
    __syncthreads();

    v8f oa[8];
#pragma unroll
    for (int nt = 0; nt < 8; ++nt)
#pragma unroll
      for (int j = 0; j < 8; ++j) oa[nt][j] = 0.0f;
#pragma unroll
    for (int kk = 0; kk < 4; ++kk) {
      v16bf a = ld_a(Pb + kk * 32, wave * 16 + lc, 136, khalf);
#pragma unroll
      for (int nt = 0; nt < 8; ++nt) {
        v16bf b = ld_b(Vs + kk * 32, nt * 16 + lc, 136, khalf);
        oa[nt] = wmma_bf16(a, b, oa[nt]);
      }
    }
#pragma unroll
    for (int nt = 0; nt < 8; ++nt)
#pragma unroll
      for (int j = 0; j < 8; ++j) {
        const int r = wave * 16 + j + 8 * khalf;
        Op[(rowBase + r) * D_MODEL + ntile * 128 + nt * 16 + lc] = f2bf(oa[nt][j]);
      }
    __syncthreads();
  }
}

// ---------------------------------------------------------------------------
// Host launcher. Inputs (setup_inputs order):
//   0 key, 1 value, 2 query, 3 Wk, 4 bk, 5 Wq, 6 bq, 7 Wv, 8 bv
// Workspace layout (bytes):
//   [0,96MB)   : Kb/Qb/Vb bf16 inputs; later reused for O (attention output)
//   [96,102MB) : WkT/WqT/WvT bf16 [N][K]
//   [102,198MB): Kp/Qp/Vp projections bf16 [16384][1024]
// ---------------------------------------------------------------------------
extern "C" void kernel_launch(void* const* d_in, const int* in_sizes, int n_in,
                              void* d_out, int out_size, void* d_ws, size_t ws_size,
                              hipStream_t stream)
{
  (void)in_sizes; (void)n_in; (void)out_size; (void)ws_size;
  const float* key   = (const float*)d_in[0];
  const float* value = (const float*)d_in[1];
  const float* query = (const float*)d_in[2];
  const float* Wk    = (const float*)d_in[3];
  const float* bk    = (const float*)d_in[4];
  const float* Wq    = (const float*)d_in[5];
  const float* bq    = (const float*)d_in[6];
  const float* Wv    = (const float*)d_in[7];
  const float* bv    = (const float*)d_in[8];
  float* out = (float*)d_out;

  const int M = MTOT, N = D_MODEL, K = D_MODEL;
  const size_t projBytes = (size_t)M * N * sizeof(__bf16); // 32 MB
  const size_t wBytes    = (size_t)N * K * sizeof(__bf16); // 2 MB
  char* ws = (char*)d_ws;

  __bf16* Kb  = (__bf16*)(ws + 0 * projBytes);
  __bf16* Qb  = (__bf16*)(ws + 1 * projBytes);
  __bf16* Vb  = (__bf16*)(ws + 2 * projBytes);
  __bf16* Ob  = (__bf16*)(ws + 0 * projBytes);            // reuse Kb region
  __bf16* WkT = (__bf16*)(ws + 3 * projBytes + 0 * wBytes);
  __bf16* WqT = (__bf16*)(ws + 3 * projBytes + 1 * wBytes);
  __bf16* WvT = (__bf16*)(ws + 3 * projBytes + 2 * wBytes);
  __bf16* Kp  = (__bf16*)(ws + 3 * projBytes + 3 * wBytes + 0 * projBytes);
  __bf16* Qp  = (__bf16*)(ws + 3 * projBytes + 3 * wBytes + 1 * projBytes);
  __bf16* Vp  = (__bf16*)(ws + 3 * projBytes + 3 * wBytes + 2 * projBytes);

  dim3 blk(256);

  // 1) convert inputs fp32 -> bf16
  const int nElem = M * N;              // 16M
  dim3 cgrid(nElem / (256 * 8));
  hipLaunchKernelGGL(cvt_bf16_kernel, cgrid, blk, 0, stream, key,   Kb, nElem);
  hipLaunchKernelGGL(cvt_bf16_kernel, cgrid, blk, 0, stream, query, Qb, nElem);
  hipLaunchKernelGGL(cvt_bf16_kernel, cgrid, blk, 0, stream, value, Vb, nElem);

  // 2) transpose weights to bf16 [N][K]
  dim3 tgrid(K / 32, N / 32);
  hipLaunchKernelGGL(transpose_w_kernel, tgrid, blk, 0, stream, Wk, WkT, N);
  hipLaunchKernelGGL(transpose_w_kernel, tgrid, blk, 0, stream, Wq, WqT, N);
  hipLaunchKernelGGL(transpose_w_kernel, tgrid, blk, 0, stream, Wv, WvT, N);

  // 3) QKV projections (bf16 -> bf16)
  dim3 ggrid(M / 128, N / 128);
  const size_t gemmLds = 40960;         // 2x (As+Bs)
  hipLaunchKernelGGL((gemm_bias_bf16<false>), ggrid, blk, gemmLds, stream, Kb, WkT, bk, (void*)Kp, M, N, K);
  hipLaunchKernelGGL((gemm_bias_bf16<false>), ggrid, blk, gemmLds, stream, Qb, WqT, bq, (void*)Qp, M, N, K);
  hipLaunchKernelGGL((gemm_bias_bf16<false>), ggrid, blk, gemmLds, stream, Vb, WvT, bv, (void*)Vp, M, N, K);

  // 4) attention per (b,h) chunk
  const size_t attnLds = 106496 + (size_t)128 * 136 * sizeof(__bf16); // 141312 B
  hipLaunchKernelGGL(attn_kernel, dim3(BB * N_HEADS), blk, attnLds, stream, Qp, Kp, Vp, Ob);

  // 5) output projection with Wv/bv (bf16 in -> fp32 out), per reference
  hipLaunchKernelGGL((gemm_bias_bf16<true>), ggrid, blk, gemmLds, stream, Ob, WvT, bv, (void*)out, M, N, K);
}
